// NemotronMoE_76390288326765
// MI455X (gfx1250) — matrix operand order repaired and checked by reading, not compiled, not run on hardware
//
#include <hip/hip_runtime.h>
#include <hip/hip_bf16.h>
#include <math.h>

// ---------------- problem constants ----------------
#define N_TOK   4096      // B*T
#define DDIM    1024
#define HDIM    4096
#define NEXP    8
#define TOPK    2
#define NENT    (N_TOK * TOPK)        // 8192 routed entries

// ---------------- GEMM tiling ----------------
#define BM      128
#define BN      128
#define BK      32
#define ASTR    40                    // padded LDS row stride (halves): 80B -> conflict-free b128 frag reads
#define BSTR    40
#define MT_ROUTED 72                  // 9216/128 max routed M-tiles
#define MT_SHARED 32                  // 4096/128
#define MT_TOTAL  (MT_ROUTED + MT_SHARED)
#define NPAD    (MT_ROUTED * BM)      // 9216 padded routed rows
#define HROWS   (NPAD + N_TOK)        // hidden-buffer rows (routed + shared)

typedef __attribute__((ext_vector_type(16))) __bf16 v16bf;
typedef __attribute__((ext_vector_type(8)))  float  v8f;
typedef __attribute__((ext_vector_type(4)))  unsigned int u32x4;
typedef __attribute__((ext_vector_type(8)))  int i32x8;
typedef __attribute__((ext_vector_type(4)))  int i32x4;

union Frag {
    v16bf v;
    uint4 q[2];
};

__device__ __forceinline__ unsigned short f2bf(float f) {
    unsigned int u = __float_as_uint(f);
    unsigned int r = u + 0x7FFFu + ((u >> 16) & 1u);   // RNE
    return (unsigned short)(r >> 16);
}

// 16B-chunk XOR swizzle for the B tile (K dimension), keyed on n so the
// transpose scatter-stores spread across banks while frag reads stay b128.
__device__ __forceinline__ int bswz(int n, int chunk) {
    int s = (n + (n >> 4)) & 3;
    return ((chunk ^ s) << 3);
}

// ---------------- TDM support (guarded) ----------------
#if defined(__has_builtin)
#if __has_builtin(__builtin_amdgcn_tensor_load_to_lds)
#define HAVE_TDM 1
#endif
#endif
#ifndef HAVE_TDM
#define HAVE_TDM 0
#endif

__device__ __forceinline__ void wait_tensor0() {
#if defined(__has_builtin) && __has_builtin(__builtin_amdgcn_s_wait_tensorcnt)
    __builtin_amdgcn_s_wait_tensorcnt(0);
#else
    asm volatile("s_wait_tensorcnt 0" ::: "memory");
#endif
}

#if HAVE_TDM
// Copy a 128x32-element bf16 tile (rows stride HDIM elems) into LDS with a
// 16B pad every 64B so it lands in the ASTR=40-half padded layout.
__device__ __forceinline__ void tdm_tile_bf16(unsigned lds_addr,
                                              const unsigned short* gsrc) {
    unsigned long long ga = (unsigned long long)(uintptr_t)gsrc;
    u32x4 g0;
    g0[0] = 1u;                                   // count=1 (valid descriptor)
    g0[1] = lds_addr;                             // lds_addr[31:0]
    g0[2] = (unsigned)(ga & 0xFFFFFFFFu);         // global_addr[95:64]
    g0[3] = (unsigned)((ga >> 32) & 0x01FFFFFFu)  // global_addr[120:96]
          | (2u << 30);                           // type=2 ("image")
    i32x8 g1;
    g1[0] = 0x06D10000;   // data_size=2B | pad_enable | pad_interval=3 (64B) | pad_amount=3 (16B)
    g1[1] = 0x10000000;   // tensor_dim0 = 4096  (bits 63:48 = 0x1000)
    g1[2] = 0x40000000;   // tensor_dim1 = 16384 (bits 95:80 = 0x4000)
    g1[3] = 0x00200000;   // tile_dim0 = 32      (bits 127:112 = 0x20)
    g1[4] = 0x00000080;   // tile_dim1 = 128, tile_dim2 = 0
    g1[5] = 4096;         // tensor_dim0_stride = 4096 elements
    g1[6] = 0;
    g1[7] = 0;
    i32x4 z4 = {0, 0, 0, 0};
#if defined(__clang_major__) && (__clang_major__ >= 23)
    i32x8 z8 = {0, 0, 0, 0, 0, 0, 0, 0};
    __builtin_amdgcn_tensor_load_to_lds(g0, g1, z4, z4, z8, 0);
#else
    __builtin_amdgcn_tensor_load_to_lds(g0, g1, z4, z4, 0);
#endif
}
#endif

// ---------------------------------------------------------------
// zero output
// ---------------------------------------------------------------
__global__ void moe_zero(float* __restrict__ o, int n) {
    int i = blockIdx.x * 256 + threadIdx.x;
    if (i < n) o[i] = 0.0f;
}

// ---------------------------------------------------------------
// router: one wave per token. logits = x @ W_router; sigmoid; top-2.
// ---------------------------------------------------------------
__global__ __launch_bounds__(256)
void moe_router(const float* __restrict__ x, const float* __restrict__ Wr,
                int* __restrict__ eids, float* __restrict__ wts) {
    int gtid = blockIdx.x * 256 + threadIdx.x;
    int tok  = gtid >> 5;
    int lane = threadIdx.x & 31;
    if (tok >= N_TOK) return;

    const float* xr = x + (size_t)tok * DDIM;
    float acc[NEXP];
#pragma unroll
    for (int e = 0; e < NEXP; ++e) acc[e] = 0.0f;

    for (int d = lane; d < DDIM; d += 32) {
        float xv = xr[d];
        const float* w = Wr + (size_t)d * NEXP;
#pragma unroll
        for (int e = 0; e < NEXP; ++e) acc[e] += xv * w[e];
    }
#pragma unroll
    for (int off = 16; off > 0; off >>= 1) {
#pragma unroll
        for (int e = 0; e < NEXP; ++e)
            acc[e] += __shfl_xor(acc[e], off, 32);
    }
    if (lane == 0) {
        float p[NEXP];
#pragma unroll
        for (int e = 0; e < NEXP; ++e) p[e] = 1.0f / (1.0f + __expf(-acc[e]));
        int i0 = 0;
#pragma unroll
        for (int e = 1; e < NEXP; ++e) if (p[e] > p[i0]) i0 = e;
        int i1 = -1;
#pragma unroll
        for (int e = 0; e < NEXP; ++e)
            if (e != i0 && (i1 < 0 || p[e] > p[i1])) i1 = e;
        float s = p[i0] + p[i1] + 1e-6f;
        eids[tok * 2 + 0] = i0;  wts[tok * 2 + 0] = p[i0] / s;
        eids[tok * 2 + 1] = i1;  wts[tok * 2 + 1] = p[i1] / s;
    }
}

// ---------------------------------------------------------------
// scatter: deterministic stable counting sort into per-expert segments
// padded to BM; builds tile tables. Single block of 256 threads.
// ---------------------------------------------------------------
__global__ __launch_bounds__(256)
void moe_scatter(const int* __restrict__ eids, const float* __restrict__ wts,
                 int* __restrict__ perm_tok, float* __restrict__ perm_w,
                 int* __restrict__ tileExpert, int* __restrict__ tileRow) {
    __shared__ unsigned char ec[NENT];
    __shared__ int cnt[256][NEXP];
    __shared__ int chunkOff[256][NEXP];
    __shared__ int segStart[NEXP], total[NEXP], padded[NEXP];

    int tid = threadIdx.x;
    for (int i = tid; i < NENT; i += 256) ec[i] = (unsigned char)eids[i];
    __syncthreads();

    int base = tid * 32;
    int c[NEXP];
#pragma unroll
    for (int e = 0; e < NEXP; ++e) c[e] = 0;
    for (int j = 0; j < 32; ++j) c[ec[base + j]]++;
#pragma unroll
    for (int e = 0; e < NEXP; ++e) cnt[tid][e] = c[e];
    __syncthreads();

    if (tid < NEXP) {
        int run = 0;
        for (int ch = 0; ch < 256; ++ch) {
            chunkOff[ch][tid] = run;
            run += cnt[ch][tid];
        }
        total[tid] = run;
    }
    __syncthreads();

    if (tid == 0) {
        int b = 0, t = 0;
        for (int e = 0; e < NEXP; ++e) {
            segStart[e] = b;
            int pad = ((total[e] + BM - 1) / BM) * BM;
            padded[e] = pad;
            for (int r = 0; r < pad / BM; ++r) {
                tileExpert[t] = e;
                tileRow[t]    = b + r * BM;
                ++t;
            }
            b += pad;
        }
        for (; t < MT_ROUTED; ++t) { tileExpert[t] = -1; tileRow[t] = 0; }
    }
    __syncthreads();

    int run[NEXP];
#pragma unroll
    for (int e = 0; e < NEXP; ++e) run[e] = segStart[e] + chunkOff[tid][e];
    for (int j = 0; j < 32; ++j) {
        int ent = base + j;
        int e = ec[ent];
        int pos = run[e]++;
        perm_tok[pos] = ent >> 1;       // token index
        perm_w[pos]   = wts[ent];
    }
    for (int e = 0; e < NEXP; ++e)
        for (int p = total[e] + tid; p < padded[e]; p += 256) {
            perm_tok[segStart[e] + p] = 0;
            perm_w[segStart[e] + p]   = 0.0f;
        }
}

// ---------------------------------------------------------------
// up grouped GEMM: h = sq_relu( Xg[BM,1024] @ W[1024,HDIM] ) -> bf16 hbuf
// grid = (HDIM/BN, MT_TOTAL), 256 threads (8 waves).
// Double-buffered LDS, one barrier per K-step, reg-prefetch pipeline.
// ---------------------------------------------------------------
__global__ __launch_bounds__(256)
void moe_up(const float* __restrict__ x,
            const float* __restrict__ W_up,      // [E][D][H]
            const float* __restrict__ W_sh_up,   // [D][H]
            const int* __restrict__ perm_tok,
            const int* __restrict__ tileExpert,
            const int* __restrict__ tileRow,
            unsigned short* __restrict__ hbuf) { // [HROWS][H] bf16
    __shared__ __align__(16) unsigned short As[2][BM * ASTR];
    __shared__ __align__(16) unsigned short Bs[2][BN * BSTR];
    __shared__ int rowIdx[BM];

    int tid = threadIdx.x;
    int ty  = blockIdx.y;
    const float* W;
    int hbase;
    if (ty < MT_ROUTED) {
        int e = tileExpert[ty];
        if (e < 0) return;                       // uniform per block
        int rbase = tileRow[ty];
        W = W_up + (size_t)e * DDIM * HDIM;
        hbase = rbase;
        if (tid < BM) rowIdx[tid] = perm_tok[rbase + tid];
    } else {
        int s = ty - MT_ROUTED;
        W = W_sh_up;
        hbase = NPAD + s * BM;
        if (tid < BM) rowIdx[tid] = s * BM + tid;
    }
    __syncthreads();

    int lane   = tid & 31;
    int wv     = tid >> 5;
    int rowSub = wv >> 1;                        // 0..3
    int colSub = wv & 1;                         // 0..1
    int n0     = blockIdx.x * BN;

    int arow  = tid >> 1;                        // 0..127
    int akoff = (tid & 1) * 16;
    const float* aSrc = x + (size_t)rowIdx[arow] * DDIM + akoff;

    int bk    = tid >> 3;                        // 0..31
    int bnoff = (tid & 7) * 16;
    const float* bSrc = W + (size_t)bk * HDIM + n0 + bnoff;

    v8f acc[2][4];
#pragma unroll
    for (int mi = 0; mi < 2; ++mi)
#pragma unroll
        for (int ni = 0; ni < 4; ++ni)
#pragma unroll
            for (int r = 0; r < 8; ++r) acc[mi][ni][r] = 0.0f;

    int klo_a = 16 * (lane >> 4);
    int klo_b = 16 * (lane >> 4);
    int c0    = klo_b >> 3;                      // 0 or 2
    int mlane = lane & 15;

    float4 ra[4], rb[4];
    auto gload = [&](int k0) {
        const float4* ap = (const float4*)(aSrc + k0);
        ra[0] = ap[0]; ra[1] = ap[1]; ra[2] = ap[2]; ra[3] = ap[3];
        const float4* bp = (const float4*)(bSrc + (size_t)k0 * HDIM);
        rb[0] = bp[0]; rb[1] = bp[1]; rb[2] = bp[2]; rb[3] = bp[3];
    };
    auto lstore = [&](int buf) {
        unsigned short* pa = &As[buf][arow * ASTR + akoff];
        float av[16] = {ra[0].x, ra[0].y, ra[0].z, ra[0].w,
                        ra[1].x, ra[1].y, ra[1].z, ra[1].w,
                        ra[2].x, ra[2].y, ra[2].z, ra[2].w,
                        ra[3].x, ra[3].y, ra[3].z, ra[3].w};
#pragma unroll
        for (int j = 0; j < 16; ++j) pa[j] = f2bf(av[j]);
        float bv[16] = {rb[0].x, rb[0].y, rb[0].z, rb[0].w,
                        rb[1].x, rb[1].y, rb[1].z, rb[1].w,
                        rb[2].x, rb[2].y, rb[2].z, rb[2].w,
                        rb[3].x, rb[3].y, rb[3].z, rb[3].w};
#pragma unroll
        for (int j = 0; j < 16; ++j) {
            int n = bnoff + j;
            Bs[buf][n * BSTR + (bk & 7) + bswz(n, bk >> 3)] = f2bf(bv[j]);
        }
    };
    auto compute = [&](int buf) {
        const unsigned short* Asb = &As[buf][0];
        const unsigned short* Bsb = &Bs[buf][0];
        Frag af[2];
#pragma unroll
        for (int mi = 0; mi < 2; ++mi) {
            int m = (rowSub * 32 + mi * 16 + mlane) * ASTR;
            af[mi].q[0] = *(const uint4*)&Asb[m + klo_a];
            af[mi].q[1] = *(const uint4*)&Asb[m + klo_a + 8];
        }
#pragma unroll
        for (int ni = 0; ni < 4; ++ni) {
            int n = colSub * 64 + ni * 16 + mlane;
            Frag bf;
            bf.q[0] = *(const uint4*)&Bsb[n * BSTR + bswz(n, c0)];
            bf.q[1] = *(const uint4*)&Bsb[n * BSTR + bswz(n, c0 + 1)];
#pragma unroll
            for (int mi = 0; mi < 2; ++mi)
                acc[mi][ni] = __builtin_amdgcn_wmma_f32_16x16x32_bf16(
                    false, af[mi].v, false, bf.v, (short)0, acc[mi][ni],
                    false, false);
        }
    };

    gload(0);
    lstore(0);
    __syncthreads();
    const int NT = DDIM / BK;
    for (int it = 0; it < NT; ++it) {
        int cur = it & 1;
        if (it + 1 < NT) {
            gload((it + 1) * BK);
            lstore(cur ^ 1);
        }
        compute(cur);
        __syncthreads();
    }

    // epilogue: sq_relu -> bf16 hbuf
    int lh = lane >> 4;
#pragma unroll
    for (int mi = 0; mi < 2; ++mi)
#pragma unroll
        for (int ni = 0; ni < 4; ++ni)
#pragma unroll
            for (int r = 0; r < 8; ++r) {
                int mloc = rowSub * 32 + mi * 16 + 8 * lh + r;
                int ngl  = n0 + colSub * 64 + ni * 16 + mlane;
                float t = acc[mi][ni][r];
                t = t > 0.0f ? t * t : 0.0f;
                hbuf[(size_t)(hbase + mloc) * HDIM + ngl] = f2bf(t);
            }
}

// ---------------------------------------------------------------
// down grouped GEMM: out += w * ( hbuf[BM,H] @ W_down[H,D] )
// grid = (DDIM/BN, MT_TOTAL), 256 threads.
// A tile (pure bf16 copy) streamed by the Tensor Data Mover when available.
// ---------------------------------------------------------------
__global__ __launch_bounds__(256)
void moe_down(const unsigned short* __restrict__ hbuf,   // [HROWS][H] bf16
              const float* __restrict__ W_down,          // [E][H][D]
              const float* __restrict__ W_sh_down,       // [H][D]
              const int* __restrict__ perm_tok,
              const float* __restrict__ perm_w,
              const int* __restrict__ tileExpert,
              const int* __restrict__ tileRow,
              float* __restrict__ out) {
    __shared__ __align__(16) unsigned short As[2][BM * ASTR];
    __shared__ __align__(16) unsigned short Bs[2][BN * BSTR];
    __shared__ int   tokRow[BM];
    __shared__ float wRow[BM];

    int tid = threadIdx.x;
    int ty  = blockIdx.y;
    const float* W;
    int hbase;
    if (ty < MT_ROUTED) {
        int e = tileExpert[ty];
        if (e < 0) return;
        int rbase = tileRow[ty];
        W = W_down + (size_t)e * HDIM * DDIM;
        hbase = rbase;
        if (tid < BM) {
            tokRow[tid] = perm_tok[rbase + tid];
            wRow[tid]   = perm_w[rbase + tid];
        }
    } else {
        int s = ty - MT_ROUTED;
        W = W_sh_down;
        hbase = NPAD + s * BM;
        if (tid < BM) {
            tokRow[tid] = s * BM + tid;
            wRow[tid]   = 1.0f;
        }
    }
    __syncthreads();

    int lane   = tid & 31;
    int wv     = tid >> 5;
    int rowSub = wv >> 1;
    int colSub = wv & 1;
    int n0     = blockIdx.x * BN;

    int arow  = tid >> 1;
    int akoff = (tid & 1) * 16;
    const unsigned short* aSrc = hbuf + (size_t)(hbase + arow) * HDIM + akoff;
    const unsigned short* aTile = hbuf + (size_t)hbase * HDIM;   // row 0 of tile

    int bk    = tid >> 3;
    int bnoff = (tid & 7) * 16;
    const float* bSrc = W + (size_t)bk * DDIM + n0 + bnoff;

    v8f acc[2][4];
#pragma unroll
    for (int mi = 0; mi < 2; ++mi)
#pragma unroll
        for (int ni = 0; ni < 4; ++ni)
#pragma unroll
            for (int r = 0; r < 8; ++r) acc[mi][ni][r] = 0.0f;

    int klo_a = 16 * (lane >> 4);
    int klo_b = 16 * (lane >> 4);
    int c0    = klo_b >> 3;
    int mlane = lane & 15;

    uint4  aq[2];
    float4 rb[4];
#if !HAVE_TDM
    auto aload = [&](int k0) {
        const uint4* ap = (const uint4*)(aSrc + k0);
        aq[0] = ap[0]; aq[1] = ap[1];
    };
    auto astore = [&](int buf) {
        *(uint4*)&As[buf][arow * ASTR + akoff]     = aq[0];
        *(uint4*)&As[buf][arow * ASTR + akoff + 8] = aq[1];
    };
#else
    (void)aq; (void)aSrc;
#endif
    auto bload = [&](int k0) {
        const float4* bp = (const float4*)(bSrc + (size_t)k0 * DDIM);
        rb[0] = bp[0]; rb[1] = bp[1]; rb[2] = bp[2]; rb[3] = bp[3];
    };
    auto bstore = [&](int buf) {
        float bv[16] = {rb[0].x, rb[0].y, rb[0].z, rb[0].w,
                        rb[1].x, rb[1].y, rb[1].z, rb[1].w,
                        rb[2].x, rb[2].y, rb[2].z, rb[2].w,
                        rb[3].x, rb[3].y, rb[3].z, rb[3].w};
#pragma unroll
        for (int j = 0; j < 16; ++j) {
            int n = bnoff + j;
            Bs[buf][n * BSTR + (bk & 7) + bswz(n, bk >> 3)] = f2bf(bv[j]);
        }
    };
    auto compute = [&](int buf) {
        const unsigned short* Asb = &As[buf][0];
        const unsigned short* Bsb = &Bs[buf][0];
        Frag af[2];
#pragma unroll
        for (int mi = 0; mi < 2; ++mi) {
            int m = (rowSub * 32 + mi * 16 + mlane) * ASTR;
            af[mi].q[0] = *(const uint4*)&Asb[m + klo_a];
            af[mi].q[1] = *(const uint4*)&Asb[m + klo_a + 8];
        }
#pragma unroll
        for (int ni = 0; ni < 4; ++ni) {
            int n = colSub * 64 + ni * 16 + mlane;
            Frag bf;
            bf.q[0] = *(const uint4*)&Bsb[n * BSTR + bswz(n, c0)];
            bf.q[1] = *(const uint4*)&Bsb[n * BSTR + bswz(n, c0 + 1)];
#pragma unroll
            for (int mi = 0; mi < 2; ++mi)
                acc[mi][ni] = __builtin_amdgcn_wmma_f32_16x16x32_bf16(
                    false, af[mi].v, false, bf.v, (short)0, acc[mi][ni],
                    false, false);
        }
    };

    const int NT = HDIM / BK;
#if HAVE_TDM
    if (wv == 0)
        tdm_tile_bf16((unsigned)(uintptr_t)&As[0][0], aTile + 0);
#else
    (void)aTile;
    aload(0); astore(0);
#endif
    bload(0); bstore(0);
#if HAVE_TDM
    if (wv == 0) wait_tensor0();
#endif
    __syncthreads();

    for (int it = 0; it < NT; ++it) {
        int cur = it & 1;
        if (it + 1 < NT) {
#if HAVE_TDM
            if (wv == 0)
                tdm_tile_bf16((unsigned)(uintptr_t)&As[cur ^ 1][0],
                              aTile + (it + 1) * BK);
#else
            aload((it + 1) * BK); astore(cur ^ 1);
#endif
            bload((it + 1) * BK); bstore(cur ^ 1);
        }
        compute(cur);
#if HAVE_TDM
        if ((it + 1 < NT) && wv == 0) wait_tensor0();
#endif
        __syncthreads();
    }

    // epilogue: weighted atomic accumulate into out
    int lh = lane >> 4;
#pragma unroll
    for (int mi = 0; mi < 2; ++mi)
#pragma unroll
        for (int ni = 0; ni < 4; ++ni)
#pragma unroll
            for (int r = 0; r < 8; ++r) {
                int mloc = rowSub * 32 + mi * 16 + 8 * lh + r;
                int ngl  = n0 + colSub * 64 + ni * 16 + mlane;
                float w = wRow[mloc];
                if (w != 0.0f)
                    atomicAdd(out + (size_t)tokRow[mloc] * DDIM + ngl,
                              w * acc[mi][ni][r]);
            }
}

// ---------------------------------------------------------------
// launch
// ---------------------------------------------------------------
extern "C" void kernel_launch(void* const* d_in, const int* in_sizes, int n_in,
                              void* d_out, int out_size, void* d_ws, size_t ws_size,
                              hipStream_t stream) {
    (void)in_sizes; (void)n_in; (void)out_size;
    const float* x    = (const float*)d_in[0];   // [N,D]
    const float* Wr   = (const float*)d_in[1];   // [D,E]
    const float* Wup  = (const float*)d_in[2];   // [E,D,H]
    const float* Wdn  = (const float*)d_in[3];   // [E,H,D]
    const float* Wsu  = (const float*)d_in[4];   // [D,H]
    const float* Wsd  = (const float*)d_in[5];   // [H,D]
    float* out = (float*)d_out;

    char* ws = (char*)d_ws;
    size_t off = 0;
    int*   eids      = (int*)  (ws + off); off += (size_t)NENT * 4;
    float* wts       = (float*)(ws + off); off += (size_t)NENT * 4;
    int*   perm_tok  = (int*)  (ws + off); off += (size_t)NPAD * 4;
    float* perm_w    = (float*)(ws + off); off += (size_t)NPAD * 4;
    int*   tileExp   = (int*)  (ws + off); off += 1024;
    int*   tileRow   = (int*)  (ws + off); off += 1024;
    off = (off + 255) & ~(size_t)255;
    unsigned short* hbuf = (unsigned short*)(ws + off);
    off += (size_t)HROWS * HDIM * 2;                 // ~104 MB
    if (off > ws_size) return;   // workspace too small: bail (no UB)

    moe_zero<<<(N_TOK * DDIM + 255) / 256, 256, 0, stream>>>(out, N_TOK * DDIM);
    moe_router<<<(N_TOK * 32 + 255) / 256, 256, 0, stream>>>(x, Wr, eids, wts);
    moe_scatter<<<1, 256, 0, stream>>>(eids, wts, perm_tok, perm_w, tileExp, tileRow);
    moe_up<<<dim3(HDIM / BN, MT_TOTAL), 256, 0, stream>>>(
        x, Wup, Wsu, perm_tok, tileExp, tileRow, hbuf);
    moe_down<<<dim3(DDIM / BN, MT_TOTAL), 256, 0, stream>>>(
        hbuf, Wdn, Wsd, perm_tok, perm_w, tileExp, tileRow, out);
}
